// LFQAutoEncoder_67860483276879
// MI455X (gfx1250) — matrix operand order repaired and checked by reading, not compiled, not run on hardware
//
#include <hip/hip_runtime.h>
#include <hip/hip_bf16.h>
#include <math.h>

// ---------------------------------------------------------------------------
// LFQ autoencoder forward for gfx1250 (MI455X).
//   z = x @ w_enc + b_enc            (bf16 WMMA, f32 accum)
//   q = sign(z); indices; entropy/commit aux
//   out = q @ w_dec + b_dec          (q rebuilt from indices, bf16 WMMA)
// ---------------------------------------------------------------------------

typedef __bf16 bf16x2  __attribute__((ext_vector_type(2)));
typedef __bf16 bf16x8  __attribute__((ext_vector_type(8)));
typedef __bf16 bf16x16 __attribute__((ext_vector_type(16)));
typedef float  f32x8   __attribute__((ext_vector_type(8)));
typedef unsigned int u32x4 __attribute__((ext_vector_type(4)));
typedef int    i32x8   __attribute__((ext_vector_type(8)));
typedef int    i32x4   __attribute__((ext_vector_type(4)));

constexpr int NB  = 16384;   // batch
constexpr int IN  = 784;     // input dim
constexpr int HID = 256;     // hidden dim
constexpr int NC  = 32;      // codebooks
constexpr int CD  = 8;       // bits per codebook
constexpr int CS  = 256;     // codebook size

#define HAVE_TDM __has_builtin(__builtin_amdgcn_tensor_load_to_lds)

// Native transcendentals: v_exp_f32 computes 2^x, v_log_f32 computes log2(x).
__device__ __forceinline__ float fast_exp2(float x) {
#if __has_builtin(__builtin_amdgcn_exp2f)
  return __builtin_amdgcn_exp2f(x);
#else
  return exp2f(x);
#endif
}
__device__ __forceinline__ float fast_log2(float x) {
#if __has_builtin(__builtin_amdgcn_logf)
  return __builtin_amdgcn_logf(x);
#else
  return log2f(x);
#endif
}

// 1-D contiguous DMA of `nelem` DWORDs from global -> LDS via the Tensor Data
// Mover (D# per CDNA5 ISA). Call from ONE wave only.
__device__ __forceinline__ void tdm_load_1d(unsigned lds_off, const void* gptr,
                                            unsigned nelem /* <= 65535 */) {
#if HAVE_TDM
  unsigned long long ga = (unsigned long long)gptr;
  u32x4 g0;
  g0[0] = 1u;                                   // count=1, user mode
  g0[1] = lds_off;                              // lds_addr (bytes)
  g0[2] = (unsigned)(ga & 0xFFFFFFFFull);       // global_addr[31:0]
  g0[3] = (unsigned)((ga >> 32) & 0x1FFFFFFull) // global_addr[56:32]
          | (2u << 30);                         // type=2 ("image")
  i32x8 g1;
  g1[0] = (int)(2u << 16);                      // data_size=4B
  g1[1] = (int)((nelem & 0xFFFFu) << 16);       // tensor_dim0[15:0]
  g1[2] = (int)((nelem >> 16) | (1u << 16));    // tensor_dim0 hi | tensor_dim1=1
  g1[3] = (int)((nelem & 0xFFFFu) << 16);       // tile_dim0
  g1[4] = 1;                                    // tile_dim1=1
  g1[5] = (int)nelem;                           // tensor_dim0_stride
  g1[6] = 0;
  g1[7] = 0;
  i32x4 gz4 = {0, 0, 0, 0};
  i32x8 gz8 = {0, 0, 0, 0, 0, 0, 0, 0};
  __builtin_amdgcn_tensor_load_to_lds(g0, g1, gz4, gz4, gz8, 0);
  __builtin_amdgcn_s_wait_tensorcnt(0);
#endif
}

// ---------------------------------------------------------------------------
// Kernel 0: zero the scratch accumulators.
// ---------------------------------------------------------------------------
__global__ void lfq_zero_ws(float* __restrict__ p, int n) {
  int i = blockIdx.x * 256 + threadIdx.x;
  if (i < n) p[i] = 0.0f;
}

// ---------------------------------------------------------------------------
// Kernel 1: fused encode GEMM + LFQ.
// Block = 512 threads (16 waves); block owns 16 rows, wave w owns N-tile w.
// LDS (dynamic, 91648 B):
//   [0,50176)      xs   : x tile 16x784 f32 (TDM)  | phase2: zs 16K + pacc 32K
//   [50176,75264)  xsb  : bf16 copy of x tile 16x784
//   [75264,91648)  wslT : w_enc K-slice, TRANSPOSED [n][k] 256x32 bf16
// ---------------------------------------------------------------------------
__global__ void __launch_bounds__(512, 1)
lfq_encode(const float* __restrict__ x, const float* __restrict__ w_enc,
           const float* __restrict__ b_enc, int* __restrict__ indices_out,
           float* __restrict__ gprob, float* __restrict__ gsums) {
  extern __shared__ char smem[];
  float*  xs   = (float*)smem;                       // 16*784 f32
  __bf16* xsb  = (__bf16*)(smem + 50176);            // 16*784 bf16
  __bf16* wslT = (__bf16*)(smem + 75264);            // 256*32 bf16
  float*  zs   = (float*)smem;                       // 16*256 f32 (alias xs)
  float*  pacc = (float*)(smem + 16 * HID * 4);      // 32*256 f32 (alias xs)

  const int tid  = threadIdx.x;
  const int lane = tid & 31;
  const int wave = tid >> 5;
  const int row0 = blockIdx.x * 16;

  // --- stage x rows into LDS (16*784 contiguous DWORDs) via TDM ------------
#if HAVE_TDM
  if (tid < 32) {
    tdm_load_1d((unsigned)(unsigned long long)(void*)xs,
                x + (size_t)row0 * IN, 16u * IN);
  }
  __syncthreads();
  // convert f32 tile -> bf16 copy (vectorized: float2 load, bf16x2 store)
  {
    const float2* xs2 = (const float2*)xs;
    bf16x2*       xb2 = (bf16x2*)xsb;
    for (int p = tid; p < 16 * IN / 2; p += 512) {
      float2 f = xs2[p];
      bf16x2 v;
      v[0] = (__bf16)f.x;
      v[1] = (__bf16)f.y;
      xb2[p] = v;
    }
  }
#else
  for (int p = tid; p < 16 * IN / 2; p += 512) {
    float2 f = ((const float2*)(x + (size_t)row0 * IN))[p];
    bf16x2 v;
    v[0] = (__bf16)f.x;
    v[1] = (__bf16)f.y;
    ((bf16x2*)xsb)[p] = v;
  }
#endif

  // --- WMMA GEMM: z[16x256] = x[16x784] * w_enc[784x256] -------------------
  const int  n0   = wave * 16;
  const int  mrow = lane & 15;
  const bool hi   = lane >= 16;
  const int  sn   = tid & 255;   // staging: column n
  const int  shal = tid >> 8;    // staging: k half (0/1)
  f32x8 acc = {0.f, 0.f, 0.f, 0.f, 0.f, 0.f, 0.f, 0.f};

  for (int kt = 0; kt < 25; ++kt) {               // 784 = 24*32 + 16
    const int k0 = kt * 32;
    __syncthreads();                              // wslT WAR (+ xsb ready @0)
    {                                             // stage w_enc -> wslT [n][k]
      const int kb = k0 + shal * 16;              // half-chunk all-valid or not
      bf16x16 v;
      if (kb < IN) {                              // 784 % 16 == 0
#pragma unroll
        for (int j = 0; j < 16; ++j)              // coalesced across sn
          v[j] = (__bf16)w_enc[(kb + j) * HID + sn];
      } else {
#pragma unroll
        for (int j = 0; j < 16; ++j) v[j] = (__bf16)0.0f;
      }
      *(bf16x16*)&wslT[sn * 32 + shal * 16] = v;  // one 32-B LDS store
    }
    __syncthreads();

    // A fragment: two contiguous 16-B LDS loads (zero upper half on tail)
    const int ka  = k0 + (hi ? 8 : 0);
    const int ka2 = k0 + (hi ? 24 : 16);
    bf16x16 a;
    bf16x8 alo = *(const bf16x8*)&xsb[mrow * IN + ka];
#pragma unroll
    for (int h = 0; h < 8; ++h) a[h] = alo[h];
    if (kt < 24) {
      bf16x8 ahi = *(const bf16x8*)&xsb[mrow * IN + ka2];
#pragma unroll
      for (int h = 0; h < 8; ++h) a[h + 8] = ahi[h];
    } else {
#pragma unroll
      for (int h = 0; h < 8; ++h) a[h + 8] = (__bf16)0.0f;
    }
    // B fragment: two 16-B LDS loads (k 8h..8h+7, 16+8h..) from transposed slice
    bf16x8 blo = *(const bf16x8*)&wslT[(n0 + mrow) * 32 + (hi ? 8 : 0)];
    bf16x8 bhi = *(const bf16x8*)&wslT[(n0 + mrow) * 32 + (hi ? 24 : 16)];
    bf16x16 b;
#pragma unroll
    for (int h = 0; h < 8; ++h) { b[h] = blo[h]; b[h + 8] = bhi[h]; }

    acc = __builtin_amdgcn_wmma_f32_16x16x32_bf16(false, a, false, b,
                                                  (short)0, acc, false, false);
  }

  __syncthreads();  // all xs/xsb reads done before aliased zs/pacc writes

  {                                               // bias + z tile to LDS
    const int   ncol = n0 + mrow;
    const float bias = b_enc[ncol];
#pragma unroll
    for (int r = 0; r < 8; ++r)
      zs[(r + (hi ? 8 : 0)) * HID + ncol] = acc[r] + bias;
  }
  for (int i = tid; i < NC * CS; i += 512) pacc[i] = 0.0f;
  __syncthreads();

  // --- LFQ: 512 threads = 16 rows x 32 codebooks ---------------------------
  {
    const int row = tid >> 5;
    const int c   = tid & 31;
    float zd[CD];
#pragma unroll
    for (int d = 0; d < CD; ++d) zd[d] = zs[row * HID + c * CD + d];

    int   idx = 0;
    float commit = 0.0f, sumz = 0.0f, suma = 0.0f;
    float stp[CD];
    const float K2 = 200.0f * 1.44269504088896f;  // 2*inv_temp * log2(e)
#pragma unroll
    for (int d = 0; d < CD; ++d) {
      bool  pos = zd[d] > 0.0f;
      idx |= (pos ? 1 : 0) << (7 - d);            // MSB-first bit order
      float df = zd[d] - (pos ? 1.0f : -1.0f);
      commit += df * df;
      sumz += zd[d];
      suma += fabsf(zd[d]);                       // = exact max logit / 200
      stp[d] = 2.0f * K2 * zd[d];
    }
    indices_out[(size_t)(row0 + row) * NC + c] = idx;

    // Softmax over all 256 sign patterns, Gray-code walk (fully unrolled so
    // ctz/g/signs constant-fold). t2 = (logit - max) in log2 units.
    const float t0 = (-sumz - suma) * K2;
    float t2 = t0;
    float S = fast_exp2(t2), T2 = S * t2;
    unsigned g = 0;
#pragma unroll
    for (int j = 1; j < CS; ++j) {
      const int b = __builtin_ctz((unsigned)j);
      g ^= 1u << b;
      const float s = stp[7 - b];
      t2 = ((g >> b) & 1u) ? (t2 + s) : (t2 - s);
      float e = fast_exp2(t2);
      S += e;
      T2 += e * t2;
    }
    const float LN2  = 0.69314718055994531f;
    const float invS = 1.0f / S;
    const float H    = LN2 * (fast_log2(S) - T2 * invS);  // per-sample entropy

    // pass 2: accumulate probabilities (skip negligible ones)
    t2 = t0;
    g  = 0;
    {
      float p = fast_exp2(t2) * invS;
      if (p > 1e-12f) atomicAdd(&pacc[c * CS + 0], p);
    }
#pragma unroll
    for (int j = 1; j < CS; ++j) {
      const int b = __builtin_ctz((unsigned)j);
      g ^= 1u << b;
      const float s = stp[7 - b];
      t2 = ((g >> b) & 1u) ? (t2 + s) : (t2 - s);
      float p = fast_exp2(t2) * invS;
      if (p > 1e-12f) atomicAdd(&pacc[c * CS + g], p);
    }

    // wave32 reduction of entropy / commit sums
    float hc = H, cc = commit;
    for (int off = 16; off > 0; off >>= 1) {
      hc += __shfl_down(hc, off, 32);
      cc += __shfl_down(cc, off, 32);
    }
    if (lane == 0) {
      atomicAdd(&gsums[0], hc);
      atomicAdd(&gsums[1], cc);
    }
  }
  __syncthreads();

  for (int i = tid; i < NC * CS; i += 512) atomicAdd(&gprob[i], pacc[i]);
}

// ---------------------------------------------------------------------------
// Kernel 2: decode GEMM, out = q @ w_dec + b_dec, q rebuilt from index bits.
// Block = 256 threads (8 waves): 16 rows x 128 cols. K = 256 = 8 chunks.
// LDS: idxs 2048 | wdlT (transposed [n][k] 128x32 bf16) 8192 | outs 8192
// ---------------------------------------------------------------------------
__global__ void __launch_bounds__(256, 1)
lfq_decode(const int* __restrict__ indices, const float* __restrict__ w_dec,
           const float* __restrict__ b_dec, float* __restrict__ out) {
  extern __shared__ char smem[];
  int*    idxs = (int*)smem;                    // 16*32 ints
  __bf16* wdlT = (__bf16*)(smem + 2048);        // 128*32 bf16
  float*  outs = (float*)(smem + 2048 + 8192);  // 16*128 f32

  const int tid   = threadIdx.x;
  const int lane  = tid & 31;
  const int wave  = tid >> 5;
  const int row0  = blockIdx.x * 16;
  const int nbase = blockIdx.y * 128;

  // stage the 16x32 index tile (512 contiguous DWORDs) via TDM
#if HAVE_TDM
  if (tid < 32) {
    tdm_load_1d((unsigned)(unsigned long long)(void*)idxs,
                indices + (size_t)row0 * NC, 16u * NC);
  }
#else
  for (int i = tid; i < 16 * NC; i += 256)
    idxs[i] = indices[(size_t)row0 * NC + i];
#endif

  const int  mrow = lane & 15;
  const bool hi   = lane >= 16;
  const int  snl  = tid & 127;   // staging: local column
  const int  shal = tid >> 7;    // staging: k half (0/1)
  f32x8 acc = {0.f, 0.f, 0.f, 0.f, 0.f, 0.f, 0.f, 0.f};
  const __bf16 pone = (__bf16)1.0f, mone = (__bf16)-1.0f;

  for (int kt = 0; kt < 8; ++kt) {
    const int k0 = kt * 32;
    __syncthreads();                            // wdlT WAR (+ idxs ready @0)
    {                                           // stage w_dec -> wdlT [n][k]
      const int kb = k0 + shal * 16;
      const int n  = nbase + snl;
      bf16x16 v;
      if (n < IN) {
#pragma unroll
        for (int j = 0; j < 16; ++j)            // coalesced across snl
          v[j] = (__bf16)w_dec[(kb + j) * IN + n];
      } else {
#pragma unroll
        for (int j = 0; j < 16; ++j) v[j] = (__bf16)0.0f;
      }
      *(bf16x16*)&wdlT[snl * 32 + shal * 16] = v;
    }
    __syncthreads();

    // A from index bits: each 8-aligned K group = one codebook's 8 bits
    const int i0 = idxs[mrow * NC + ((k0 + (hi ? 8 : 0)) >> 3)];
    const int i1 = idxs[mrow * NC + ((k0 + (hi ? 24 : 16)) >> 3)];
    bf16x16 a;
#pragma unroll
    for (int h = 0; h < 8; ++h) {
      a[h]     = ((i0 >> (7 - h)) & 1) ? pone : mone;
      a[h + 8] = ((i1 >> (7 - h)) & 1) ? pone : mone;
    }
    bf16x8 blo = *(const bf16x8*)&wdlT[(wave * 16 + mrow) * 32 + (hi ? 8 : 0)];
    bf16x8 bhi = *(const bf16x8*)&wdlT[(wave * 16 + mrow) * 32 + (hi ? 24 : 16)];
    bf16x16 b;
#pragma unroll
    for (int h = 0; h < 8; ++h) { b[h] = blo[h]; b[h + 8] = bhi[h]; }

    acc = __builtin_amdgcn_wmma_f32_16x16x32_bf16(false, a, false, b,
                                                  (short)0, acc, false, false);
  }

  // epilogue: bias + LDS bounce for coalesced 32-B global stores
  {
    const int   ncl  = wave * 16 + mrow;
    const int   n    = nbase + ncl;
    const float bias = (n < IN) ? b_dec[n] : 0.0f;
#pragma unroll
    for (int r = 0; r < 8; ++r)
      outs[(r + (hi ? 8 : 0)) * 128 + ncl] = acc[r] + bias;
  }
  __syncthreads();
  {
    const int row = tid >> 4;            // 0..15
    const int c8  = (tid & 15) * 8;      // 0..120
    const int n   = nbase + c8;
    if (n + 7 < IN) {                    // 784-768=16: 8-groups all-or-nothing
      float4 v0 = *(const float4*)&outs[row * 128 + c8];
      float4 v1 = *(const float4*)&outs[row * 128 + c8 + 4];
      float* dst = &out[(size_t)(row0 + row) * IN + n];
      *(float4*)dst       = v0;
      *(float4*)(dst + 4) = v1;
    }
  }
}

// ---------------------------------------------------------------------------
// Kernel 3: finalize aux scalar.
// aux = 0.1 * (per_sample_entropy - codebook_entropy) + 0.25 * commit
// ---------------------------------------------------------------------------
__global__ void __launch_bounds__(256, 1)
lfq_finalize(const float* __restrict__ gprob, const float* __restrict__ gsums,
             float* __restrict__ aux_out) {
  __shared__ float red[256];
  const int tid = threadIdx.x;
  float s = 0.0f;
  const float LN2 = 0.69314718055994531f;
  for (int i = tid; i < NC * CS; i += 256) {
    float p = gprob[i] * (1.0f / (float)NB);    // avg_prob
    s += -p * (fast_log2(fmaxf(p, 1e-5f)) * LN2);
  }
  red[tid] = s;
  __syncthreads();
  for (int k = 128; k > 0; k >>= 1) {
    if (tid < k) red[tid] += red[tid + k];
    __syncthreads();
  }
  if (tid == 0) {
    float cb_ent  = red[0] / (float)NC;
    float ps_ent  = gsums[0] / ((float)NB * (float)NC);
    float commitv = gsums[1] / ((float)NB * (float)HID);
    aux_out[0] = 0.1f * (ps_ent - 1.0f * cb_ent) + 0.25f * commitv;
  }
}

// ---------------------------------------------------------------------------
extern "C" void kernel_launch(void* const* d_in, const int* in_sizes, int n_in,
                              void* d_out, int out_size, void* d_ws,
                              size_t ws_size, hipStream_t stream) {
  const float* x     = (const float*)d_in[0];
  const float* w_enc = (const float*)d_in[1];
  const float* b_enc = (const float*)d_in[2];
  const float* w_dec = (const float*)d_in[3];
  const float* b_dec = (const float*)d_in[4];

  float* out     = (float*)d_out;                         // 16384*784
  int*   indices = (int*)(out + (size_t)NB * IN);         // 16384*32
  float* aux     = (float*)(out + (size_t)NB * IN + (size_t)NB * NC);

  float* gprob = (float*)d_ws;        // 32*256 prob sums
  float* gsums = gprob + NC * CS;     // [entropy_sum, commit_sum]

  lfq_zero_ws<<<(NC * CS + 2 + 255) / 256, 256, 0, stream>>>(gprob,
                                                             NC * CS + 2);

  const size_t smem_enc = 50176 + 25088 + 16384;          // 91648 B
  (void)hipFuncSetAttribute((const void*)lfq_encode,
                            hipFuncAttributeMaxDynamicSharedMemorySize,
                            (int)smem_enc);
  lfq_encode<<<NB / 16, 512, smem_enc, stream>>>(x, w_enc, b_enc, indices,
                                                 gprob, gsums);

  const size_t smem_dec = 2048 + 8192 + 8192;             // 18432 B
  dim3 gdec(NB / 16, (IN + 127) / 128);
  lfq_decode<<<gdec, 256, smem_dec, stream>>>(indices, w_dec, b_dec, out);

  lfq_finalize<<<1, 256, 0, stream>>>(gprob, gsums, aux);
}
